// MaskedSelfAttention_43181601194937
// MI455X (gfx1250) — matrix operand order repaired and checked by reading, not compiled, not run on hardware
//
#include <hip/hip_runtime.h>
#include <hip/hip_bf16.h>

typedef __attribute__((ext_vector_type(16))) _Float16 v16h;
typedef __attribute__((ext_vector_type(8)))  _Float16 v8h;
typedef __attribute__((ext_vector_type(8)))  float    v8f;

static constexpr int BATCH = 4;
static constexpr int SEQ   = 4096;
static constexpr int DIM   = 1024;

static __device__ __forceinline__ v8f wmma_f16(v16h a, v16h b, v8f c) {
  // D = A(16x32 f16) * B(32x16 f16) + C(16x16 f32)
  return __builtin_amdgcn_wmma_f32_16x16x32_f16(false, a, false, b, (short)0, c, false, false);
}

static __device__ __forceinline__ v16h pack2(uint4 lo, uint4 hi) {
  union { v16h v; uint4 q[2]; } u;
  u.q[0] = lo; u.q[1] = hi;
  return u.v;
}

// ---------------------------------------------------------------------------
// Kernel 0: fp32 -> f16 conversion (8 elements per thread, b128 in / b128 out)
// ---------------------------------------------------------------------------
__global__ __launch_bounds__(256) void cvt_f16_kernel(
    const float* __restrict__ in, _Float16* __restrict__ out, int n8)
{
  int i = blockIdx.x * blockDim.x + threadIdx.x;
  if (i >= n8) return;
  const float4* p = (const float4*)in + (size_t)i * 2;
  float4 a = p[0], b = p[1];
  v8h h;
  h[0]=(_Float16)a.x; h[1]=(_Float16)a.y; h[2]=(_Float16)a.z; h[3]=(_Float16)a.w;
  h[4]=(_Float16)b.x; h[5]=(_Float16)b.y; h[6]=(_Float16)b.z; h[7]=(_Float16)b.w;
  *((v8h*)out + i) = h;
}

// ---------------------------------------------------------------------------
// Kernel 1: Q/K/V = x @ W^T + b from f16-staged inputs.
// One wave -> 32 rows x 64 cols: 8 WMMAs per 32-wide K step, 12 b128 loads.
// V is written transposed ([B, D, T]).
// ---------------------------------------------------------------------------
__global__ __launch_bounds__(256) void qkv_proj_kernel(
    const _Float16* __restrict__ xh,
    const _Float16* __restrict__ Wqh, const float* __restrict__ bq,
    const _Float16* __restrict__ Wkh, const float* __restrict__ bk,
    const _Float16* __restrict__ Wvh, const float* __restrict__ bv,
    _Float16* __restrict__ Qh, _Float16* __restrict__ Kh,
    _Float16* __restrict__ Vth)
{
  const int lane = threadIdx.x & 31;
  const int widx = blockIdx.x * (blockDim.x >> 5) + (threadIdx.x >> 5);

  const int MT = (BATCH * SEQ) / 32;   // 512 row tiles (32 rows each)
  const int NT = DIM / 64;             // 16 col supertiles
  const int mat  = widx / (MT * NT);   // 0=Q 1=K 2=V
  const int rem  = widx % (MT * NT);
  const int row0 = (rem / NT) * 32;
  const int col0 = (rem % NT) * 64;

  const _Float16* W = (mat == 0) ? Wqh : (mat == 1) ? Wkh : Wvh;
  const float* bias = (mat == 0) ? bq  : (mat == 1) ? bk  : bv;

  const int m16  = lane & 15;
  const int hsel = (lane < 16) ? 0 : 1;
  const int mh   = hsel * 8;

  v8f acc[2][4] = {};

  // Per-lane row pointers (fixed across the K loop)
  const _Float16* pa0 = xh + (size_t)(row0 + m16) * DIM;        // A tile 0
  const _Float16* pa1 = xh + (size_t)(row0 + 16 + m16) * DIM;   // A tile 1
  const _Float16* pb[4];
  #pragma unroll
  for (int nt = 0; nt < 4; ++nt)
    pb[nt] = W + (size_t)(col0 + nt * 16 + m16) * DIM;

  for (int kk = 0; kk < DIM; kk += 32) {
    // ---- batch ALL loads of this K-step first (12 independent b128 loads)
    uint4 la[2][2], lb[4][2];
    la[0][0] = *(const uint4*)(pa0 + kk + hsel * 16);
    la[0][1] = *(const uint4*)(pa0 + kk + hsel * 16 + 8);
    la[1][0] = *(const uint4*)(pa1 + kk + hsel * 16);
    la[1][1] = *(const uint4*)(pa1 + kk + hsel * 16 + 8);
    #pragma unroll
    for (int nt = 0; nt < 4; ++nt) {
      lb[nt][0] = *(const uint4*)(pb[nt] + kk + hsel * 16);
      lb[nt][1] = *(const uint4*)(pb[nt] + kk + hsel * 16 + 8);
    }
    // ---- then 8 WMMAs
    v16h a0 = pack2(la[0][0], la[0][1]);
    v16h a1 = pack2(la[1][0], la[1][1]);
    #pragma unroll
    for (int nt = 0; nt < 4; ++nt) {
      v16h bm = pack2(lb[nt][0], lb[nt][1]);
      acc[0][nt] = wmma_f16(a0, bm, acc[0][nt]);
      acc[1][nt] = wmma_f16(a1, bm, acc[1][nt]);
    }
  }

  // ---- epilogue: bias, f16 convert, scatter per C layout (M = i + 8*hsel)
  #pragma unroll
  for (int t = 0; t < 2; ++t) {
    #pragma unroll
    for (int nt = 0; nt < 4; ++nt) {
      const int col = col0 + nt * 16 + m16;
      const float bval = bias[col];
      #pragma unroll
      for (int i = 0; i < 8; ++i) {
        float v = acc[t][nt][i] + bval;
        int r = row0 + t * 16 + mh + i;        // global row in [0, B*T)
        if (mat == 0) {
          Qh[(size_t)r * DIM + col] = (_Float16)v;
        } else if (mat == 1) {
          Kh[(size_t)r * DIM + col] = (_Float16)v;
        } else {
          int bb = r >> 12;                    // r / SEQ
          int tt = r & (SEQ - 1);
          Vth[((size_t)bb * DIM + col) * SEQ + tt] = (_Float16)v;
        }
      }
    }
  }
}

// ---------------------------------------------------------------------------
// Kernel 2: flash-style causal attention.
// One 256-thread block (8 waves) per (batch, 16-query block).
// Wave w owns D slice [128w, 128w+128); Q A-tiles resident in registers.
// ---------------------------------------------------------------------------
__global__ __launch_bounds__(256) void attn_kernel(
    const _Float16* __restrict__ Qh, const _Float16* __restrict__ Kh,
    const _Float16* __restrict__ Vth, float* __restrict__ out)
{
  __shared__ float Spart[8][16][32];                  // per-wave partial scores
  __shared__ float Sm[16][32];                        // reduced + masked scores
  __shared__ __align__(16) _Float16 Pbuf[16][32];     // probabilities (f16)
  __shared__ float mrow[16], lrow[16], scalef[16];

  const int bb   = blockIdx.x >> 8;          // batch
  const int q0   = (blockIdx.x & 255) * 16;  // first query row
  const int wave = threadIdx.x >> 5;
  const int lane = threadIdx.x & 31;
  const int m16  = lane & 15;
  const int hsel = (lane < 16) ? 0 : 1;
  const int mh   = hsel * 8;
  const int dw   = wave * 128;               // this wave's D slice

  if (wave == 0 && lane < 16) { mrow[lane] = -__builtin_inff(); lrow[lane] = 0.0f; }

  // Preload Q A-tiles for this wave's D slice (reused across all key blocks)
  v16h aq[4];
  {
    const _Float16* qp = Qh + (size_t)(bb * SEQ + q0 + m16) * DIM + dw;
    uint4 lq[4][2];
    #pragma unroll
    for (int kt = 0; kt < 4; ++kt) {
      lq[kt][0] = *(const uint4*)(qp + kt * 32 + hsel * 16);
      lq[kt][1] = *(const uint4*)(qp + kt * 32 + hsel * 16 + 8);
    }
    #pragma unroll
    for (int kt = 0; kt < 4; ++kt) aq[kt] = pack2(lq[kt][0], lq[kt][1]);
  }

  v8f acc[8] = {};

  __syncthreads();

  const int klast = q0 + 15;                 // causal: keys <= query
  for (int k0 = 0; k0 <= klast; k0 += 32) {
    // ---- partial S = Q_slice @ K_slice^T : batch 8 loads per N-tile
    v8f sacc[2] = {};
    #pragma unroll
    for (int nt = 0; nt < 2; ++nt) {
      const _Float16* kp =
          Kh + (size_t)(bb * SEQ + k0 + nt * 16 + m16) * DIM + dw + hsel * 16;
      uint4 lk[4][2];
      #pragma unroll
      for (int kt = 0; kt < 4; ++kt) {
        lk[kt][0] = *(const uint4*)(kp + kt * 32);
        lk[kt][1] = *(const uint4*)(kp + kt * 32 + 8);
      }
      #pragma unroll
      for (int kt = 0; kt < 4; ++kt)
        sacc[nt] = wmma_f16(aq[kt], pack2(lk[kt][0], lk[kt][1]), sacc[nt]);
    }
    if (k0 + 32 <= klast) {   // speculative prefetch of next key block
      __builtin_prefetch(Kh + (size_t)(bb * SEQ + k0 + 32 + m16) * DIM + dw, 0, 0);
    }
    #pragma unroll
    for (int nt = 0; nt < 2; ++nt)
      #pragma unroll
      for (int i = 0; i < 8; ++i)
        Spart[wave][mh + i][nt * 16 + m16] = sacc[nt][i];
    __syncthreads();

    // ---- reduce 8 wave partials, scale by 1/sqrt(D), apply causal mask
    for (int e = threadIdx.x; e < 512; e += 256) {
      int m = e >> 5, n = e & 31;
      float s = 0.0f;
      #pragma unroll
      for (int w = 0; w < 8; ++w) s += Spart[w][m][n];
      Sm[m][n] = ((k0 + n) <= (q0 + m)) ? s * 0.03125f : -__builtin_inff();
    }
    __syncthreads();

    // ---- online softmax for this 16x32 block (wave 0, one lane per row)
    if (wave == 0 && lane < 16) {
      const int m = lane;
      float rmax = Sm[m][0];
      #pragma unroll
      for (int n = 1; n < 32; ++n) rmax = fmaxf(rmax, Sm[m][n]);
      float mold = mrow[m];
      float mnew = fmaxf(mold, rmax);
      float sc   = __expf(mold - mnew);
      float sum  = 0.0f;
      #pragma unroll
      for (int n = 0; n < 32; ++n) {
        float p = __expf(Sm[m][n] - mnew);
        Pbuf[m][n] = (_Float16)p;
        sum += p;
      }
      lrow[m]   = lrow[m] * sc + sum;
      mrow[m]   = mnew;
      scalef[m] = sc;
    }
    __syncthreads();

    // ---- rescale accumulators, then acc += P(16x32) @ V(32x128 slice)
    float sc_i[8];
    #pragma unroll
    for (int i = 0; i < 8; ++i) sc_i[i] = scalef[mh + i];
    #pragma unroll
    for (int nt = 0; nt < 8; ++nt)
      #pragma unroll
      for (int i = 0; i < 8; ++i) acc[nt][i] *= sc_i[i];

    v16h pA = pack2(*(const uint4*)&Pbuf[m16][hsel * 16],
                    *(const uint4*)&Pbuf[m16][hsel * 16 + 8]);

    // batch V loads in groups of 4 tiles (8 b128 loads -> 4 WMMAs)
    const _Float16* vbase =
        Vth + ((size_t)bb * DIM + (dw + m16)) * SEQ + k0 + hsel * 16;
    #pragma unroll
    for (int g = 0; g < 2; ++g) {
      uint4 lv[4][2];
      #pragma unroll
      for (int j = 0; j < 4; ++j) {
        const _Float16* vp = vbase + (size_t)(g * 4 + j) * 16 * SEQ;
        lv[j][0] = *(const uint4*)(vp);
        lv[j][1] = *(const uint4*)(vp + 8);
      }
      #pragma unroll
      for (int j = 0; j < 4; ++j)
        acc[g * 4 + j] = wmma_f16(pA, pack2(lv[j][0], lv[j][1]), acc[g * 4 + j]);
    }
  }

  // ---- epilogue: divide by row sums, write fp32 output
  float invl[8];
  #pragma unroll
  for (int i = 0; i < 8; ++i) invl[i] = 1.0f / lrow[mh + i];
  #pragma unroll
  for (int nt = 0; nt < 8; ++nt) {
    #pragma unroll
    for (int i = 0; i < 8; ++i) {
      out[(size_t)(bb * SEQ + q0 + mh + i) * DIM + dw + nt * 16 + m16] =
          acc[nt][i] * invl[i];
    }
  }
}

// ---------------------------------------------------------------------------
extern "C" void kernel_launch(void* const* d_in, const int* in_sizes, int n_in,
                              void* d_out, int out_size, void* d_ws, size_t ws_size,
                              hipStream_t stream) {
  const float* x  = (const float*)d_in[0];
  const float* Wq = (const float*)d_in[1];
  const float* bq = (const float*)d_in[2];
  const float* Wk = (const float*)d_in[3];
  const float* bk = (const float*)d_in[4];
  const float* Wv = (const float*)d_in[5];
  const float* bv = (const float*)d_in[6];
  float* out = (float*)d_out;

  const size_t N = (size_t)BATCH * SEQ * DIM;   // 16M elements

  // Workspace: Q | K | V^T | x(f16) | Wq(f16) | Wk(f16) | Wv(f16)
  _Float16* Qh  = (_Float16*)d_ws;
  _Float16* Kh  = Qh + N;
  _Float16* Vth = Kh + N;
  _Float16* xh  = Vth + N;
  _Float16* Wqh = xh + N;
  _Float16* Wkh = Wqh + (size_t)DIM * DIM;
  _Float16* Wvh = Wkh + (size_t)DIM * DIM;

  // Stage inputs as f16
  cvt_f16_kernel<<<(int)(N / 8 / 256), 256, 0, stream>>>(x, xh, (int)(N / 8));
  const int w8 = DIM * DIM / 8;
  cvt_f16_kernel<<<w8 / 256, 256, 0, stream>>>(Wq, Wqh, w8);
  cvt_f16_kernel<<<w8 / 256, 256, 0, stream>>>(Wk, Wkh, w8);
  cvt_f16_kernel<<<w8 / 256, 256, 0, stream>>>(Wv, Wvh, w8);

  const int nwaves = ((BATCH * SEQ) / 32) * (DIM / 64) * 3;   // 24576
  qkv_proj_kernel<<<nwaves / 8, 256, 0, stream>>>(xh, Wqh, bq, Wkh, bk, Wvh, bv,
                                                  Qh, Kh, Vth);
  attn_kernel<<<BATCH * (SEQ / 16), 256, 0, stream>>>(Qh, Kh, Vth, out);
}
